// ResidualGatedGCNLayer_23931557773478
// MI455X (gfx1250) — matrix operand order repaired and checked by reading, not compiled, not run on hardware
//
#include <hip/hip_runtime.h>
#include <cstdint>

#define BDIM 8
#define NDIM 256
#define HDIM 64
#define NODE_ROWS (BDIM * NDIM)          /* 2048 */
#define R_EDGE    (BDIM * NDIM * NDIM)   /* 524288 */
#define EPS_BN    1e-5f

typedef __attribute__((ext_vector_type(16))) __bf16 v16bf;
typedef __attribute__((ext_vector_type(4)))  __bf16 bf4;
typedef __attribute__((ext_vector_type(8)))  float  v8f;
typedef __attribute__((ext_vector_type(4)))  float  f4;

__device__ __forceinline__ v8f wmma_bf16(v16bf a, v16bf b, v8f c) {
    // emits v_wmma_f32_16x16x32_bf16
    return __builtin_amdgcn_wmma_f32_16x16x32_bf16(
        false, a, false, b, (short)0, c, false, false);
}

// B fragment (32x16, K x N): lanes 0-15 hold K=K0..K0+15 (elem i=K0+i),
// lanes 16-31 hold K=K0+16..31.  B[k][n] = W[n][k] -> 16 contiguous floats of row n.
__device__ __forceinline__ v16bf load_wfrag(const float* __restrict__ W, int n, int kbase) {
    v16bf u;
#pragma unroll
    for (int i = 0; i < 16; ++i) u[i] = (__bf16)W[n * HDIM + kbase + i];
    return u;
}

// Swizzled-LDS destination for element (jl, h) of a 16x64 tile so that each
// (kc, khalf, row) A-fragment is 16 contiguous bf16 (one aligned 32B load).
//   K = kc*32 + g*16 + khalf*8 + i   <->  lane half g, frag elem index = khalf*8 + i
__device__ __forceinline__ int swz_base(int h) {
    const int kc    = h >> 5;
    const int krem  = h & 31;
    const int khalf = (krem >> 3) & 1;
    const int g     = (krem >> 4) & 1;
    const int i     = krem & 7;
    return ((kc * 2 + g) * 16) * 16 + khalf * 8 + i;   // + row*16 added by caller
}

// ---------------- K1: node linears -> VxeT/VxnT (transposed [h][node]) + Ux ---------------
__global__ void __launch_bounds__(64)
node_linear_kernel(const float* __restrict__ x,
                   const float* __restrict__ Ve_W, const float* __restrict__ Ve_b,
                   const float* __restrict__ Un_W, const float* __restrict__ Un_b,
                   const float* __restrict__ Vn_W, const float* __restrict__ Vn_b,
                   float* __restrict__ VxeT, float* __restrict__ Ux,
                   float* __restrict__ VxnT, float* __restrict__ stats) {
    __shared__ float xr[HDIM];
    const int row = blockIdx.x;
    const int o   = threadIdx.x;
    xr[o] = x[row * HDIM + o];
    __syncthreads();
    float a = 0.f, b = 0.f, c = 0.f;
#pragma unroll 8
    for (int h = 0; h < HDIM; ++h) {
        float xv = xr[h];
        a += xv * Ve_W[o * HDIM + h];
        b += xv * Un_W[o * HDIM + h];
        c += xv * Vn_W[o * HDIM + h];
    }
    VxeT[o * NODE_ROWS + row] = a + Ve_b[o];   // transposed: j contiguous per channel
    Ux  [row * HDIM + o]      = b + Un_b[o];
    VxnT[o * NODE_ROWS + row] = c + Vn_b[o];
    if (row == 0) {
        for (int k = o; k < 256; k += HDIM) stats[k] = 0.f;
    }
}

// ---------------- K2: edge pass 1 — BN stats + gated node aggregation ----------------
__global__ void __launch_bounds__(256)
edge_pass1_kernel(const float* __restrict__ e, const float* __restrict__ Ue_W,
                  const float* __restrict__ Ue_b, const float* __restrict__ VxeT,
                  const float* __restrict__ Ux,  const float* __restrict__ VxnT,
                  float* __restrict__ x_tmp, float* __restrict__ stats) {
    __shared__ __align__(32) __bf16 eh[8][NDIM * 4];   // 8 waves x swizzled 16x64 bf16
    __shared__ float xacc[HDIM], esum[HDIM], esq[HDIM];

    const int w    = blockIdx.x;            // b*N + i
    const int b    = w >> 8;
    const int tid  = threadIdx.x;
    const int wv   = tid >> 5, lane = tid & 31;
    const int nsub = lane & 15, khalf = lane >> 4;

    if (tid < HDIM) { xacc[tid] = 0.f; esum[tid] = 0.f; esq[tid] = 0.f; }

    v16bf Bf[4][2];
    float cb[4];
#pragma unroll
    for (int nt = 0; nt < 4; ++nt) {
        const int o = nt * 16 + nsub;
#pragma unroll
        for (int kc = 0; kc < 2; ++kc)
            Bf[nt][kc] = load_wfrag(Ue_W, o, kc * 32 + khalf * 16);
        cb[nt] = Ue_b[o] + VxeT[o * NODE_ROWS + w];
    }

    float agg[4] = {0, 0, 0, 0}, se[4] = {0, 0, 0, 0}, se2[4] = {0, 0, 0, 0};
    const float* eslab = e + (size_t)w * (NDIM * HDIM);
    const int hbase = (lane * 4) & 63;                    // constant per lane across q
    const int c0    = swz_base(hbase);

    for (int pass = 0; pass < 2; ++pass) {
        const int j0 = pass * 128 + wv * 16;
        const float* tp = eslab + j0 * HDIM;
#pragma unroll
        for (int q = 0; q < 8; ++q) {                     // coalesced f32 -> swizzled bf16
            f4 v = *(const f4*)(tp + (lane + q * 32) * 4);
            bf4 hv; hv.x = (__bf16)v.x; hv.y = (__bf16)v.y;
                    hv.z = (__bf16)v.z; hv.w = (__bf16)v.w;
            const int jl = (lane >> 4) + 2 * q;
            *(bf4*)&eh[wv][c0 + jl * 16] = hv;
        }
        __syncthreads();

        v16bf Af[2];
#pragma unroll
        for (int kc = 0; kc < 2; ++kc)                    // one aligned 32B load each
            Af[kc] = *(const v16bf*)&eh[wv][((kc * 2 + khalf) * 16 + nsub) * 16];

#pragma unroll
        for (int nt = 0; nt < 4; ++nt) {
            const int o = nt * 16 + nsub;
            v8f c = {cb[nt], cb[nt], cb[nt], cb[nt], cb[nt], cb[nt], cb[nt], cb[nt]};
            c = wmma_bf16(Af[0], Bf[nt][0], c);
            c = wmma_bf16(Af[1], Bf[nt][1], c);
            const float* vjp = VxeT + o * NODE_ROWS + b * NDIM + j0 + khalf * 8;
            const float* vnp = VxnT + o * NODE_ROWS + b * NDIM + j0 + khalf * 8;
            f4 vj0 = *(const f4*)vjp, vj1 = *(const f4*)(vjp + 4);
            f4 vn0 = *(const f4*)vnp, vn1 = *(const f4*)(vnp + 4);
#pragma unroll
            for (int r = 0; r < 8; ++r) {
                const float vj = (r < 4) ? vj0[r & 3] : vj1[r & 3];
                const float vn = (r < 4) ? vn0[r & 3] : vn1[r & 3];
                const float et = c[r] + vj;
                const float g  = __builtin_amdgcn_rcpf(1.f + __expf(-et));
                agg[nt] = fmaf(g, vn, agg[nt]);
                se[nt] += et;
                se2[nt] = fmaf(et, et, se2[nt]);
            }
        }
        __syncthreads();
    }

#pragma unroll
    for (int nt = 0; nt < 4; ++nt) {                      // LDS reduction (ds_add_f32)
        const int o = nt * 16 + nsub;
        atomicAdd(&xacc[o], agg[nt]);
        atomicAdd(&esum[o], se[nt]);
        atomicAdd(&esq[o],  se2[nt]);
    }
    __syncthreads();

    if (tid < HDIM) {
        const float xv = Ux[w * HDIM + tid] + xacc[tid];
        x_tmp[w * HDIM + tid] = xv;
        unsafeAtomicAdd(&stats[tid],       xv);           // global_atomic_add_f32
        unsafeAtomicAdd(&stats[64 + tid],  xv * xv);
        unsafeAtomicAdd(&stats[128 + tid], esum[tid]);
        unsafeAtomicAdd(&stats[192 + tid], esq[tid]);
    }
}

// ---------------- K3: fold BN stats into affine coefs ----------------
__global__ void __launch_bounds__(64)
bn_coef_kernel(const float* __restrict__ stats,
               const float* __restrict__ bne_g, const float* __restrict__ bne_b,
               const float* __restrict__ bnn_g, const float* __restrict__ bnn_b,
               float* __restrict__ coef) {
    const int o = threadIdx.x;
    const float mx = stats[o] / (float)NODE_ROWS;
    const float vx = stats[64 + o] / (float)NODE_ROWS - mx * mx;
    const float ax = bnn_g[o] * rsqrtf(vx + EPS_BN);
    coef[o]      = ax;
    coef[64 + o] = bnn_b[o] - mx * ax;
    const float me = stats[128 + o] / (float)R_EDGE;
    const float ve = stats[192 + o] / (float)R_EDGE - me * me;
    const float ae = bne_g[o] * rsqrtf(ve + EPS_BN);
    coef[128 + o] = ae;
    coef[192 + o] = bne_b[o] - me * ae;
}

// ---------------- K4x: node finalize ----------------
__global__ void __launch_bounds__(256)
node_final_kernel(const float* __restrict__ x, const float* __restrict__ x_tmp,
                  const float* __restrict__ coef, float* __restrict__ out) {
    const int idx = blockIdx.x * 256 + threadIdx.x;
    const int o   = idx & (HDIM - 1);
    float v = coef[o] * x_tmp[idx] + coef[64 + o];
    v = v > 0.f ? v : 0.f;
    out[idx] = v + x[idx];
}

// ---------------- K4e: edge pass 2 — recompute, BN+ReLU+residual, store ----------------
__global__ void __launch_bounds__(256)
edge_pass2_kernel(const float* __restrict__ e, const float* __restrict__ Ue_W,
                  const float* __restrict__ Ue_b, const float* __restrict__ VxeT,
                  const float* __restrict__ coef, float* __restrict__ eout) {
    __shared__ __align__(32) __bf16 eh[8][NDIM * 4];   // swizzled bf16 tiles (WMMA A)
    __shared__ __align__(16) float  ef[8][NDIM * 4];   // f32 out-transpose tiles

    const int w    = blockIdx.x;
    const int b    = w >> 8;
    const int tid  = threadIdx.x;
    const int wv   = tid >> 5, lane = tid & 31;
    const int nsub = lane & 15, khalf = lane >> 4;

    v16bf Bf[4][2];
    float cb[4], ae[4], ce[4];
#pragma unroll
    for (int nt = 0; nt < 4; ++nt) {
        const int o = nt * 16 + nsub;
#pragma unroll
        for (int kc = 0; kc < 2; ++kc)
            Bf[nt][kc] = load_wfrag(Ue_W, o, kc * 32 + khalf * 16);
        cb[nt] = Ue_b[o] + VxeT[o * NODE_ROWS + w];
        ae[nt] = coef[128 + o];
        ce[nt] = coef[192 + o];
    }

    const float* eslab = e    + (size_t)w * (NDIM * HDIM);
    float*       oslab = eout + (size_t)w * (NDIM * HDIM);
    const int hbase = (lane * 4) & 63;
    const int c0    = swz_base(hbase);

    for (int pass = 0; pass < 2; ++pass) {
        const int j0 = pass * 128 + wv * 16;
        const float* tp = eslab + j0 * HDIM;
        f4 esav[8];                                      // residual kept in registers
#pragma unroll
        for (int q = 0; q < 8; ++q) {
            f4 v = *(const f4*)(tp + (lane + q * 32) * 4);
            esav[q] = v;
            bf4 hv; hv.x = (__bf16)v.x; hv.y = (__bf16)v.y;
                    hv.z = (__bf16)v.z; hv.w = (__bf16)v.w;
            const int jl = (lane >> 4) + 2 * q;
            *(bf4*)&eh[wv][c0 + jl * 16] = hv;
        }
        __syncthreads();

        v16bf Af[2];
#pragma unroll
        for (int kc = 0; kc < 2; ++kc)
            Af[kc] = *(const v16bf*)&eh[wv][((kc * 2 + khalf) * 16 + nsub) * 16];

#pragma unroll
        for (int nt = 0; nt < 4; ++nt) {
            const int o = nt * 16 + nsub;
            v8f c = {cb[nt], cb[nt], cb[nt], cb[nt], cb[nt], cb[nt], cb[nt], cb[nt]};
            c = wmma_bf16(Af[0], Bf[nt][0], c);
            c = wmma_bf16(Af[1], Bf[nt][1], c);
            const float* vjp = VxeT + o * NODE_ROWS + b * NDIM + j0 + khalf * 8;
            f4 vj0 = *(const f4*)vjp, vj1 = *(const f4*)(vjp + 4);
#pragma unroll
            for (int r = 0; r < 8; ++r) {
                const float vj = (r < 4) ? vj0[r & 3] : vj1[r & 3];
                const float et = c[r] + vj;
                float en = fmaf(ae[nt], et, ce[nt]);
                en = en > 0.f ? en : 0.f;
                ef[wv][(khalf * 8 + r) * HDIM + o] = en; // BN+ReLU result (no residual)
            }
        }
        __syncthreads();

        float* op = oslab + j0 * HDIM;
#pragma unroll
        for (int q = 0; q < 8; ++q) {                    // coalesced: += residual, store
            const int base = (lane + q * 32) * 4;
            f4 r = *(const f4*)&ef[wv][base];
            r.x += esav[q].x; r.y += esav[q].y; r.z += esav[q].z; r.w += esav[q].w;
            *(f4*)(op + base) = r;
        }
        __syncthreads();
    }
}

// ---------------- launch ----------------
extern "C" void kernel_launch(void* const* d_in, const int* in_sizes, int n_in,
                              void* d_out, int out_size, void* d_ws, size_t ws_size,
                              hipStream_t stream) {
    (void)in_sizes; (void)n_in; (void)out_size; (void)ws_size;
    const float* x    = (const float*)d_in[0];
    const float* e    = (const float*)d_in[1];
    const float* UeW  = (const float*)d_in[2];
    const float* Ueb  = (const float*)d_in[3];
    const float* VeW  = (const float*)d_in[4];
    const float* Veb  = (const float*)d_in[5];
    const float* UnW  = (const float*)d_in[6];
    const float* Unb  = (const float*)d_in[7];
    const float* VnW  = (const float*)d_in[8];
    const float* Vnb  = (const float*)d_in[9];
    const float* bneg = (const float*)d_in[10];
    const float* bneb = (const float*)d_in[11];
    const float* bnng = (const float*)d_in[12];
    const float* bnnb = (const float*)d_in[13];

    float* ws    = (float*)d_ws;
    float* VxeT  = ws;                     // 131072 (transposed [h][node])
    float* VxnT  = ws + 131072;            // 131072 (transposed)
    float* Ux    = ws + 262144;            // 131072 (row-major)
    float* x_tmp = ws + 393216;            // 131072
    float* stats = ws + 524288;            // 256
    float* coef  = ws + 524544;            // 256

    float* x_out = (float*)d_out;
    float* e_out = (float*)d_out + NODE_ROWS * HDIM;

    node_linear_kernel<<<NODE_ROWS, 64, 0, stream>>>(
        x, VeW, Veb, UnW, Unb, VnW, Vnb, VxeT, Ux, VxnT, stats);
    edge_pass1_kernel<<<NODE_ROWS, 256, 0, stream>>>(
        e, UeW, Ueb, VxeT, Ux, VxnT, x_tmp, stats);
    bn_coef_kernel<<<1, 64, 0, stream>>>(stats, bneg, bneb, bnng, bnnb, coef);
    node_final_kernel<<<NODE_ROWS * HDIM / 256, 256, 0, stream>>>(x, x_tmp, coef, x_out);
    edge_pass2_kernel<<<NODE_ROWS, 256, 0, stream>>>(e, UeW, Ueb, VxeT, coef, e_out);
}
